// HypergraphModel_12927851561244
// MI455X (gfx1250) — matrix operand order repaired and checked by reading, not compiled, not run on hardware
//
#include <hip/hip_runtime.h>
#include <hip/hip_bf16.h>

// ---------------------------------------------------------------------------
// CDNA5 (gfx1250, wave32) implementation of the hypergraph model.
// Heavy GEMMs (adj @ X dominates: 2 x 8192x8192x128) run on v_wmma_f32_16x16x32_bf16
// with software-pipelined, register-buffered LDS staging.
// ---------------------------------------------------------------------------

#define N_NODES 100000
#define N_EDGES 8192
#define DEG     32
#define NNZ     (N_EDGES * DEG)
#define D_IN    128
#define D_HID   128
#define D_EDGE  64
#define BN_EPS  1e-5f

typedef __attribute__((ext_vector_type(16))) __bf16 bf16x16;
typedef __attribute__((ext_vector_type(8)))  float  f32x8;

// ------------------------------- utility kernels ---------------------------

__global__ void zero_f32v4(float* p, long n4) {
    long i = (long)blockIdx.x * blockDim.x + threadIdx.x;
    float4 z = make_float4(0.f, 0.f, 0.f, 0.f);
    for (; i < n4; i += (long)gridDim.x * blockDim.x)
        reinterpret_cast<float4*>(p)[i] = z;
}

__global__ void cvt_f32_to_bf16(const float* in, __bf16* out, int n) {
    int i = blockIdx.x * blockDim.x + threadIdx.x;
    if (i < n) out[i] = (__bf16)in[i];
}

// per-feature sum / sumsq over rows; blockDim.x == cols, stats zero-initialized
__global__ void col_stats(const float* __restrict__ x, int rows, int cols,
                          float* __restrict__ stats) {
    int f = threadIdx.x;
    float s = 0.f, s2 = 0.f;
    for (int r = blockIdx.x; r < rows; r += gridDim.x) {
        float v = x[(size_t)r * cols + f];
        s += v; s2 += v * v;
    }
    atomicAdd(&stats[f], s);
    atomicAdd(&stats[cols + f], s2);
}

__global__ void bn_finalize(const float* __restrict__ stats,
                            const float* __restrict__ g, const float* __restrict__ b,
                            int cols, float rows,
                            float* __restrict__ scale, float* __restrict__ shift) {
    int f = threadIdx.x;
    if (f < cols) {
        float mu  = stats[f] / rows;
        float var = stats[cols + f] / rows - mu * mu;
        float s   = g[f] * rsqrtf(var + BN_EPS);
        scale[f] = s;
        shift[f] = b[f] - mu * s;
    }
}

// out[i] = bf16(x[i]*scale[i&mask] + shift[i&mask])
__global__ void apply_bn_bf16(const float* __restrict__ x,
                              const float* __restrict__ scale,
                              const float* __restrict__ shift,
                              __bf16* __restrict__ out, int n, int mask) {
    int i = blockIdx.x * blockDim.x + threadIdx.x;
    if (i < n) {
        int f = i & mask;
        out[i] = (__bf16)(x[i] * scale[f] + shift[f]);
    }
}

// ------------------------------ sparse kernels -----------------------------

// g[e,f] = sum_{j in edge e} w[j] * (x[src[j],f]*scale[f]+shift[f]);  sw[e] = sum w[j]
// eid = j / DEG by construction -> contiguous groups, no atomics.
__global__ void edge_weighted_gather(const float* __restrict__ x,
                                     const int* __restrict__ src,
                                     const float* __restrict__ w,
                                     const float* __restrict__ scale,
                                     const float* __restrict__ shift,
                                     int useBN,
                                     __bf16* __restrict__ g,
                                     float* __restrict__ sw) {
    int e = blockIdx.x, f = threadIdx.x;           // 128 threads
    float sc = useBN ? scale[f] : 1.f;
    float sh = useBN ? shift[f] : 0.f;
    float acc = 0.f, wsum = 0.f;
    int base = e * DEG;
    #pragma unroll 4
    for (int j = 0; j < DEG; ++j) {
        float wj = w[base + j];
        int   n  = src[base + j];
        acc  += wj * (x[(size_t)n * D_HID + f] * sc + sh);
        wsum += wj;
    }
    g[(size_t)e * D_HID + f] = (__bf16)acc;
    if (f == 0 && sw) sw[e] = wsum;
}

// out[src[j],f] += w[j] * m[eid[j],f]   (edge->node scatter, atomic)
__global__ void scatter_edge_to_node(const float* __restrict__ m,
                                     const float* __restrict__ w,
                                     const int* __restrict__ src,
                                     float* __restrict__ out) {
    int e = blockIdx.x, f = threadIdx.x;
    float mv = m[(size_t)e * D_HID + f];
    int base = e * DEG;
    #pragma unroll 4
    for (int j = 0; j < DEG; ++j)
        atomicAdd(&out[(size_t)src[base + j] * D_HID + f], w[base + j] * mv);
}

// nf[e,f] = mean_{j in edge e} relu(x[src[j],f])   (cnt == DEG exactly)
__global__ void edge_relu_mean(const float* __restrict__ x,
                               const int* __restrict__ src,
                               __bf16* __restrict__ nf) {
    int e = blockIdx.x, f = threadIdx.x;
    float acc = 0.f;
    int base = e * DEG;
    #pragma unroll 4
    for (int j = 0; j < DEG; ++j)
        acc += fmaxf(x[(size_t)src[base + j] * D_HID + f], 0.f);
    nf[(size_t)e * D_HID + f] = (__bf16)(acc * (1.f / DEG));
}

// ------------------------------- WMMA GEMM ---------------------------------
// C[M,128] = A[M,K] @ B[K,128] (+epilogue). N fixed at 128. 8 waves/block.
// Wave w owns column tile w (16 cols) and 4 row tiles (BM=64).
// Software pipelined: tile k+1 fetched into registers while tile k is consumed.
#define BM 64
#define BN 128
#define BK 32
#define LDA_S 40   // padded LDS stride (bf16 elems), 80 B (16B-aligned rows)
#define LDB_S 40   // padded LDS stride for transposed B tile

template <bool A_IS_F32, int EPI>   // EPI: 0 none, 1 rowscale*bias, 2 +addend,relu, 3 relu
__global__ __launch_bounds__(256)
void gemm_wmma(const void* __restrict__ Av, const __bf16* __restrict__ B,
               float* __restrict__ outF, __bf16* __restrict__ outB,
               int K, int lda,
               const float* __restrict__ rowscale,
               const float* __restrict__ bias,
               const float* __restrict__ addend) {
    __shared__ alignas(16) __bf16 As[BM * LDA_S];
    __shared__ alignas(16) __bf16 Bs[BN * LDB_S];   // transposed: Bs[col * LDB_S + k]

    const int m0   = blockIdx.x * BM;
    const int t    = threadIdx.x;
    const int wave = t >> 5;             // 0..7 == column tile index
    const int lane = t & 31;
    const int lh   = lane >> 4;          // half-wave
    const int ll   = lane & 15;

    // staging ownership: A -> one 16B chunk; B -> 2 columns x 8 K-rows
    const int ar = t >> 2;               // 0..63 A row
    const int ac = (t & 3) * 8;          // A col start
    const int bc = (t & 63) * 2;         // B column pair
    const int bh = t >> 6;               // 0..3 -> K rows bh*8..bh*8+7

    union RegA { uint4 u; __bf16 h[8]; } aR;
    union RegB { uint4 u; unsigned short s[8]; } bR0, bR1;

    auto fetch = [&](int k0) {
        if (A_IS_F32) {
            const float* Ap = (const float*)Av + (size_t)(m0 + ar) * lda + (k0 + ac);
            float4 f0 = ((const float4*)Ap)[0];
            float4 f1 = ((const float4*)Ap)[1];
            aR.h[0] = (__bf16)f0.x; aR.h[1] = (__bf16)f0.y;
            aR.h[2] = (__bf16)f0.z; aR.h[3] = (__bf16)f0.w;
            aR.h[4] = (__bf16)f1.x; aR.h[5] = (__bf16)f1.y;
            aR.h[6] = (__bf16)f1.z; aR.h[7] = (__bf16)f1.w;
            __builtin_prefetch(Ap + BK, 0, 3);     // global_prefetch_b8: next-next tile
        } else {
            aR.u = *(const uint4*)((const __bf16*)Av + (size_t)(m0 + ar) * lda + (k0 + ac));
        }
        const __bf16* Bp = B + (size_t)(k0 + bh * 8) * BN + bc;
        #pragma unroll
        for (int r = 0; r < 8; ++r) {
            unsigned int v = *(const unsigned int*)(Bp + (size_t)r * BN);
            bR0.s[r] = (unsigned short)(v & 0xffffu);
            bR1.s[r] = (unsigned short)(v >> 16);
        }
    };
    auto stage = [&]() {
        *(uint4*)&As[ar * LDA_S + ac]            = aR.u;
        *(uint4*)&Bs[bc * LDB_S + bh * 8]        = bR0.u;
        *(uint4*)&Bs[(bc + 1) * LDB_S + bh * 8]  = bR1.u;
    };

    f32x8 acc[4] = {};                   // 4 row tiles of 16x16

    fetch(0);
    stage();
    __syncthreads();

    for (int k0 = 0; k0 < K; k0 += BK) {
        const bool more = (k0 + BK) < K;
        if (more) fetch(k0 + BK);        // overlap global latency with WMMA below

        // B fragment: lane<16 holds K {0..7,16..23} of col ll; lane>=16 holds K {8..15,24..31}
        bf16x16 bf;
        {
            const __bf16* p = &Bs[(wave * 16 + ll) * LDB_S + lh * 8];
            #pragma unroll
            for (int e = 0; e < 8; ++e) { bf[e] = p[e]; bf[e + 8] = p[e + 16]; }
        }
        #pragma unroll
        for (int tm = 0; tm < 4; ++tm) {
            // A fragment: lane<16 -> K {0..7,16..23}, lane>=16 -> K {8..15,24..31}
            const __bf16* pa = &As[(tm * 16 + ll) * LDA_S + lh * 8];
            bf16x16 af;
            #pragma unroll
            for (int e = 0; e < 8; ++e) { af[e] = pa[e]; af[e + 8] = pa[e + 16]; }
            acc[tm] = __builtin_amdgcn_wmma_f32_16x16x32_bf16(
                false, af, false, bf, (short)0, acc[tm], false, false);
        }
        if (more) {
            __syncthreads();
            stage();
            __syncthreads();
        }
    }

    // epilogue: C/D layout -> VGPR v, lane l: M = v + (l>>4)*8, N = l&15
    #pragma unroll
    for (int tm = 0; tm < 4; ++tm) {
        #pragma unroll
        for (int v = 0; v < 8; ++v) {
            int gm = m0 + tm * 16 + lh * 8 + v;
            int gn = wave * 16 + ll;
            float c = acc[tm][v];
            if (EPI == 1) c += rowscale[gm] * bias[gn];
            if (EPI == 2) { c += addend[(size_t)gm * BN + gn]; c = fmaxf(c, 0.f); }
            if (EPI == 3) c = fmaxf(c, 0.f);
            if (outF) outF[(size_t)gm * BN + gn] = c;
            if (outB) outB[(size_t)gm * BN + gn] = (__bf16)c;
        }
    }
}

// ------------------------------- classifier --------------------------------

__global__ void classifier_k(const float* __restrict__ h,
                             const float* __restrict__ w1, const float* __restrict__ b1,
                             const float* __restrict__ w2, const float* __restrict__ b2,
                             float* __restrict__ out) {
    int e = blockIdx.x * blockDim.x + threadIdx.x;
    if (e >= N_EDGES) return;
    const float* hr = h + (size_t)e * D_HID;
    float t[16];
    #pragma unroll
    for (int i = 0; i < 16; ++i) t[i] = b1[i];
    for (int k = 0; k < D_HID; ++k) {
        float hv = hr[k];
        #pragma unroll
        for (int i = 0; i < 16; ++i) t[i] += hv * w1[k * 16 + i];
    }
    #pragma unroll
    for (int i = 0; i < 16; ++i) t[i] = fmaxf(t[i], 0.f);
    #pragma unroll
    for (int o = 0; o < 2; ++o) {
        float s = b2[o];
        #pragma unroll
        for (int i = 0; i < 16; ++i) s += t[i] * w2[i * 2 + o];
        out[(size_t)e * 2 + o] = s;
    }
}

// ------------------------------- host driver -------------------------------

static inline size_t alignup(size_t o) { return (o + 255) & ~(size_t)255; }

extern "C" void kernel_launch(void* const* d_in, const int* in_sizes, int n_in,
                              void* d_out, int out_size, void* d_ws, size_t ws_size,
                              hipStream_t stream) {
    (void)in_sizes; (void)n_in; (void)out_size; (void)ws_size;

    const float* x    = (const float*)d_in[0];
    const int*   src  = (const int*)d_in[1];           // edge_index row 0 (first NNZ)
    const float* ew   = (const float*)d_in[2];
    const float* ef   = (const float*)d_in[3];
    const float* adj  = (const float*)d_in[4];
    // d_in[5] == T; fixed at 2 in setup, unrolled host-side (graph-capture safe)
    const float* n1g  = (const float*)d_in[6];
    const float* n1b  = (const float*)d_in[7];
    const float* n2g  = (const float*)d_in[8];
    const float* n2b  = (const float*)d_in[9];
    const float* W1   = (const float*)d_in[10];
    const float* b1   = (const float*)d_in[11];
    const float* W2   = (const float*)d_in[12];
    const float* b2   = (const float*)d_in[13];
    const float* Wn   = (const float*)d_in[14];
    const float* We   = (const float*)d_in[15];
    const float* Wg   = (const float*)d_in[16];
    const float* cw1  = (const float*)d_in[17];
    const float* cb1  = (const float*)d_in[18];
    const float* cw2  = (const float*)d_in[19];
    const float* cb2  = (const float*)d_in[20];
    float* out = (float*)d_out;

    // ---- workspace carve-up
    char* ws = (char*)d_ws;
    size_t off = 0;
    auto carve = [&](size_t bytes) { void* p = ws + off; off = alignup(off + bytes); return p; };
    float*  stats1 = (float*)carve(2 * D_IN   * 4);
    float*  stats2 = (float*)carve(2 * D_EDGE * 4);
    float*  scale1 = (float*)carve(D_IN * 4);
    float*  shift1 = (float*)carve(D_IN * 4);
    float*  scale2 = (float*)carve(D_EDGE * 4);
    float*  shift2 = (float*)carve(D_EDGE * 4);
    __bf16* W1b = (__bf16*)carve(D_IN   * D_HID * 2);
    __bf16* W2b = (__bf16*)carve(D_HID  * D_HID * 2);
    __bf16* Wnb = (__bf16*)carve(D_HID  * D_HID * 2);
    __bf16* Web = (__bf16*)carve(D_EDGE * D_HID * 2);
    __bf16* Wgb = (__bf16*)carve(D_HID  * D_HID * 2);
    __bf16* gbuf = (__bf16*)carve((size_t)N_EDGES * D_HID * 2);
    float*  sw   = (float*) carve((size_t)N_EDGES * 4);
    float*  mbuf = (float*) carve((size_t)N_EDGES * D_HID * 4);
    float*  xbuf = (float*) carve((size_t)N_NODES * D_HID * 4);
    __bf16* nfb  = (__bf16*)carve((size_t)N_EDGES * D_HID * 2);
    __bf16* efb  = (__bf16*)carve((size_t)N_EDGES * D_EDGE * 2);
    float*  t1   = (float*) carve((size_t)N_EDGES * D_HID * 4);
    float*  hf   = (float*) carve((size_t)N_EDGES * D_HID * 4);
    __bf16* hb   = (__bf16*)carve((size_t)N_EDGES * D_HID * 2);
    __bf16* pb   = (__bf16*)carve((size_t)N_EDGES * D_HID * 2);

    const dim3 blk256(256);
    const long xElems = (long)N_NODES * D_HID;

    // ---- batchnorm statistics (both norms)
    zero_f32v4<<<1, 256, 0, stream>>>(stats1, (2 * D_IN) / 4);
    zero_f32v4<<<1, 256, 0, stream>>>(stats2, (2 * D_EDGE) / 4);
    col_stats<<<512, D_IN,   0, stream>>>(x,  N_NODES, D_IN,   stats1);
    col_stats<<<256, D_EDGE, 0, stream>>>(ef, N_EDGES, D_EDGE, stats2);
    bn_finalize<<<1, D_IN,   0, stream>>>(stats1, n1g, n1b, D_IN,   (float)N_NODES, scale1, shift1);
    bn_finalize<<<1, D_EDGE, 0, stream>>>(stats2, n2g, n2b, D_EDGE, (float)N_EDGES, scale2, shift2);

    // ---- weights to bf16
    cvt_f32_to_bf16<<<(D_IN  * D_HID + 255) / 256, blk256, 0, stream>>>(W1, W1b, D_IN  * D_HID);
    cvt_f32_to_bf16<<<(D_HID * D_HID + 255) / 256, blk256, 0, stream>>>(W2, W2b, D_HID * D_HID);
    cvt_f32_to_bf16<<<(D_HID * D_HID + 255) / 256, blk256, 0, stream>>>(Wn, Wnb, D_HID * D_HID);
    cvt_f32_to_bf16<<<(D_EDGE * D_HID + 255) / 256, blk256, 0, stream>>>(We, Web, D_EDGE * D_HID);
    cvt_f32_to_bf16<<<(D_HID * D_HID + 255) / 256, blk256, 0, stream>>>(Wg, Wgb, D_HID * D_HID);
    apply_bn_bf16<<<(N_EDGES * D_EDGE + 255) / 256, blk256, 0, stream>>>(
        ef, scale2, shift2, efb, N_EDGES * D_EDGE, D_EDGE - 1);

    const int gemmGrid = N_EDGES / BM;   // 128 blocks, N==128 per block

    // ---- hconv1: g1 = seg(w * bn(x)[src]) ; m1 = g1@W1 + sw*b1 ; x1 = scatter(w*m1)
    zero_f32v4<<<4096, 256, 0, stream>>>(xbuf, xElems / 4);
    edge_weighted_gather<<<N_EDGES, D_HID, 0, stream>>>(x, src, ew, scale1, shift1, 1, gbuf, sw);
    gemm_wmma<false, 1><<<gemmGrid, 256, 0, stream>>>(gbuf, W1b, mbuf, nullptr,
                                                      D_IN, D_IN, sw, b1, nullptr);
    scatter_edge_to_node<<<N_EDGES, D_HID, 0, stream>>>(mbuf, ew, src, xbuf);

    // ---- hconv2 (same, no BN), then relu + per-edge mean pooling
    edge_weighted_gather<<<N_EDGES, D_HID, 0, stream>>>(xbuf, src, ew, nullptr, nullptr, 0, gbuf, sw);
    zero_f32v4<<<4096, 256, 0, stream>>>(xbuf, xElems / 4);
    gemm_wmma<false, 1><<<gemmGrid, 256, 0, stream>>>(gbuf, W2b, mbuf, nullptr,
                                                      D_HID, D_HID, sw, b2, nullptr);
    scatter_edge_to_node<<<N_EDGES, D_HID, 0, stream>>>(mbuf, ew, src, xbuf);
    edge_relu_mean<<<N_EDGES, D_HID, 0, stream>>>(xbuf, src, nfb);

    // ---- h = relu(nf@Wn + bn(ef)@We)
    gemm_wmma<false, 0><<<gemmGrid, 256, 0, stream>>>(efb, Web, t1, nullptr,
                                                      D_EDGE, D_EDGE, nullptr, nullptr, nullptr);
    gemm_wmma<false, 2><<<gemmGrid, 256, 0, stream>>>(nfb, Wnb, hf, hb,
                                                      D_HID, D_HID, nullptr, nullptr, t1);

    // ---- T = 2 propagation steps: h = relu(adj @ (h @ Wg))
    for (int t = 0; t < 2; ++t) {
        gemm_wmma<false, 0><<<gemmGrid, 256, 0, stream>>>(hb, Wgb, nullptr, pb,
                                                          D_HID, D_HID, nullptr, nullptr, nullptr);
        gemm_wmma<true, 3><<<gemmGrid, 256, 0, stream>>>(adj, pb, hf, hb,
                                                         N_EDGES, N_EDGES, nullptr, nullptr, nullptr);
    }

    // ---- classifier head
    classifier_k<<<(N_EDGES + 255) / 256, blk256, 0, stream>>>(hf, cw1, cb1, cw2, cb2, out);
}
